// myAttentionalPropagation_13932873909013
// MI455X (gfx1250) — matrix operand order repaired and checked by reading, not compiled, not run on hardware
//
#include <hip/hip_runtime.h>
#include <hip/hip_bf16.h>

// ---- problem constants (from reference setup_inputs) ----
#define B_    2
#define N_    50000
#define F_    128
#define E_    800000
#define EP_   (E_ + N_)          // edges + self loops
#define BN_   (B_ * N_)          // 100000
#define BNF_  (BN_ * F_)         // 12.8M
#define NEG_SLOPE 0.2f
#define EPS_  1e-16f
#define ROWT_ 5                  // row tiles per wave (6250 tiles = 1250 blocks * 5)

typedef __attribute__((ext_vector_type(16))) _Float16 v16h;
typedef __attribute__((ext_vector_type(8)))  float    v8f;

// monotonic float<->uint key so unsigned atomicMax == float max
__device__ __forceinline__ unsigned fkey(float f) {
    unsigned u = __float_as_uint(f);
    return (u & 0x80000000u) ? ~u : (u | 0x80000000u);
}
__device__ __forceinline__ float funkey(unsigned k) {
    unsigned u = (k & 0x80000000u) ? (k & 0x7FFFFFFFu) : ~k;
    return __uint_as_float(u);
}

__device__ __forceinline__ float lrelu(float x) {
    return x > 0.0f ? x : NEG_SLOPE * x;
}

// ---------------------------------------------------------------------------
// Kernel 1: h = X @ W  (X flattened [BN,128] f32, W [128,128] f32, h f32)
// Block = 8 waves; wave w owns column tile w (cols 16w..16w+15) and ROWT_
// consecutive 16-row tiles. W is staged once per block into LDS as f16 in
// transposed [n][k] order, so B fragments are contiguous ds_load_b128 pairs
// and are cached in VGPRs across all row tiles. K=128 -> 4 WMMAs per tile.
// ---------------------------------------------------------------------------
__global__ void __launch_bounds__(256)
gat_gemm_wmma(const float* __restrict__ X, const float* __restrict__ Wm,
              float* __restrict__ H) {
    __shared__ _Float16 WT[F_ * F_];             // [n][k], 32 KB

    // ---- cooperative transpose + f32->f16 convert of W into LDS ----
    {
        const int t = threadIdx.x;
#pragma unroll
        for (int i = 0; i < 16; ++i) {
            const int idx = (i * 256 + t) * 4;   // flat index into W (k-major)
            const int k = idx >> 7, n = idx & (F_ - 1);
            float4 w4 = *reinterpret_cast<const float4*>(Wm + idx);
            WT[(n + 0) * F_ + k] = (_Float16)w4.x;
            WT[(n + 1) * F_ + k] = (_Float16)w4.y;
            WT[(n + 2) * F_ + k] = (_Float16)w4.z;
            WT[(n + 3) * F_ + k] = (_Float16)w4.w;
        }
    }
    __syncthreads();

    const int lane    = threadIdx.x & 31;
    const int colTile = threadIdx.x >> 5;        // 0..7
    const int half_id = lane >> 4;               // 0/1 half-wave
    const int m       = lane & 15;
    const int bn      = colTile * 16 + m;        // B-fragment column

    // ---- B fragments for all 4 K-chunks, register-resident ----
    // lanes 0-15: K = K0+0..15 ; lanes 16-31: K = K0+16..31 (contiguous in WT)
    v16h bf[4];
#pragma unroll
    for (int kc = 0; kc < 4; ++kc)
        bf[kc] = *reinterpret_cast<const v16h*>(WT + bn * F_ + kc * 32 + half_id * 16);

#pragma unroll
    for (int rt = 0; rt < ROWT_; ++rt) {
        const int rowTile = blockIdx.x * ROWT_ + rt;
        const int arow    = rowTile * 16 + m;    // A-fragment row for this lane
        v8f c = {};
#pragma unroll
        for (int kc = 0; kc < 4; ++kc) {
            const int K0 = kc * 32;
            // A fragment (contiguous k order, same as bf):
            // halves 0..7  : K = K0 + half_id*16     + 0..7
            // halves 8..15 : K = K0 + half_id*16 + 8 + 0..7
            const float4* xa = reinterpret_cast<const float4*>(X + (size_t)arow * F_ + K0 + half_id * 16);
            const float4* xb = reinterpret_cast<const float4*>(X + (size_t)arow * F_ + K0 + half_id * 16 + 8);
            float4 f0 = xa[0], f1 = xa[1];
            float4 f2 = xb[0], f3 = xb[1];
            v16h a;
            a[0] = (_Float16)f0.x; a[1] = (_Float16)f0.y; a[2] = (_Float16)f0.z; a[3] = (_Float16)f0.w;
            a[4] = (_Float16)f1.x; a[5] = (_Float16)f1.y; a[6] = (_Float16)f1.z; a[7] = (_Float16)f1.w;
            a[8] = (_Float16)f2.x; a[9] = (_Float16)f2.y; a[10] = (_Float16)f2.z; a[11] = (_Float16)f2.w;
            a[12] = (_Float16)f3.x; a[13] = (_Float16)f3.y; a[14] = (_Float16)f3.z; a[15] = (_Float16)f3.w;

            c = __builtin_amdgcn_wmma_f32_16x16x32_f16(
                    /*neg_a=*/false, a, /*neg_b=*/false, bf[kc],
                    /*c_mod=*/(short)0, c, /*reuse_a=*/false, /*reuse_b=*/false);
        }
        // store D: VGPR r -> row = rowTile*16 + half_id*8 + r, col = bn
        float* hp = H + (size_t)(rowTile * 16 + half_id * 8) * F_ + bn;
#pragma unroll
        for (int r = 0; r < 8; ++r)
            hp[(size_t)r * F_] = c[r];
    }
}

// ---------------------------------------------------------------------------
// Kernel 2: per-row attention coefficients a_src/a_dst (wave per row)
// ---------------------------------------------------------------------------
__global__ void __launch_bounds__(256)
gat_dots(const float* __restrict__ H, const float* __restrict__ attS,
         const float* __restrict__ attD, float* __restrict__ asrc,
         float* __restrict__ adst) {
    const int lane = threadIdx.x & 31;
    const long long row = ((long long)blockIdx.x * blockDim.x + threadIdx.x) >> 5;
    if (row >= BN_) return;
    float4 hv = reinterpret_cast<const float4*>(H + row * F_)[lane];
    float4 sv = reinterpret_cast<const float4*>(attS)[lane];
    float4 dv = reinterpret_cast<const float4*>(attD)[lane];
    float ps = hv.x * sv.x + hv.y * sv.y + hv.z * sv.z + hv.w * sv.w;
    float pd = hv.x * dv.x + hv.y * dv.y + hv.z * dv.z + hv.w * dv.w;
#pragma unroll
    for (int off = 16; off > 0; off >>= 1) {
        ps += __shfl_xor(ps, off, 32);
        pd += __shfl_xor(pd, off, 32);
    }
    if (lane == 0) { asrc[row] = ps; adst[row] = pd; }
}

// ---------------------------------------------------------------------------
// Kernel 3: init out<-bias (fuses +bias), maxkey<-key(-inf), denom<-0
// ---------------------------------------------------------------------------
__global__ void __launch_bounds__(256)
gat_init(float* __restrict__ out, const float* __restrict__ bias,
         unsigned* __restrict__ maxkey, float* __restrict__ denom) {
    long long idx = (long long)blockIdx.x * blockDim.x + threadIdx.x;
    if (idx < (long long)BNF_) out[idx] = bias[idx & (F_ - 1)];
    if (idx < (long long)BN_) { maxkey[idx] = 0x007FFFFFu; denom[idx] = 0.0f; }
}

// helper: edge endpoints (self loops appended after E_)
__device__ __forceinline__ void edge_sd(const long long* __restrict__ ei,
                                        long long e, long long& s, long long& d) {
    if (e < E_) { s = ei[e]; d = ei[E_ + e]; }
    else        { s = d = e - E_; }
}

// ---------------------------------------------------------------------------
// Kernel 4: segment max of leaky-relu logits (thread per (batch,edge))
// ---------------------------------------------------------------------------
__global__ void __launch_bounds__(256)
gat_emax(const long long* __restrict__ ei, const float* __restrict__ asrc,
         const float* __restrict__ adst, unsigned* __restrict__ maxkey) {
    long long t = (long long)blockIdx.x * blockDim.x + threadIdx.x;
    if (t >= (long long)B_ * EP_) return;
    long long b = t / EP_, e = t % EP_, s, d;
    edge_sd(ei, e, s, d);
    float v = lrelu(asrc[b * N_ + s] + adst[b * N_ + d]);
    atomicMax(&maxkey[b * N_ + d], fkey(v));
}

// ---------------------------------------------------------------------------
// Kernel 5: segment sum of exp(logit - max) (thread per (batch,edge))
// ---------------------------------------------------------------------------
__global__ void __launch_bounds__(256)
gat_esum(const long long* __restrict__ ei, const float* __restrict__ asrc,
         const float* __restrict__ adst, const unsigned* __restrict__ maxkey,
         float* __restrict__ denom) {
    long long t = (long long)blockIdx.x * blockDim.x + threadIdx.x;
    if (t >= (long long)B_ * EP_) return;
    long long b = t / EP_, e = t % EP_, s, d;
    edge_sd(ei, e, s, d);
    float v = lrelu(asrc[b * N_ + s] + adst[b * N_ + d]);
    atomicAdd(&denom[b * N_ + d], expf(v - funkey(maxkey[b * N_ + d])));
}

// ---------------------------------------------------------------------------
// Kernel 6: scatter out[dst] += alpha * h[src]   (wave per (batch,edge);
// lane covers 4 features -> float4 gather + 4x global_atomic_add_f32)
// ---------------------------------------------------------------------------
__global__ void __launch_bounds__(256)
gat_scatter(const long long* __restrict__ ei, const float* __restrict__ H,
            const float* __restrict__ asrc, const float* __restrict__ adst,
            const unsigned* __restrict__ maxkey, const float* __restrict__ denom,
            float* __restrict__ out) {
    const int lane = threadIdx.x & 31;
    long long w = ((long long)blockIdx.x * blockDim.x + threadIdx.x) >> 5;
    if (w >= (long long)B_ * EP_) return;
    long long b = w / EP_, e = w % EP_, s, d;
    edge_sd(ei, e, s, d);
    long long ns = b * N_ + s, nd = b * N_ + d;
    float v = lrelu(asrc[ns] + adst[nd]);
    float alpha = expf(v - funkey(maxkey[nd])) / (denom[nd] + EPS_);
    float4 hv = reinterpret_cast<const float4*>(H + ns * F_)[lane];
    float* op = out + nd * F_ + lane * 4;
    atomicAdd(op + 0, hv.x * alpha);
    atomicAdd(op + 1, hv.y * alpha);
    atomicAdd(op + 2, hv.z * alpha);
    atomicAdd(op + 3, hv.w * alpha);
}

// ---------------------------------------------------------------------------
extern "C" void kernel_launch(void* const* d_in, const int* in_sizes, int n_in,
                              void* d_out, int out_size, void* d_ws, size_t ws_size,
                              hipStream_t stream) {
    const float*     x    = (const float*)d_in[0];      // [B,N,F]
    const long long* ei   = (const long long*)d_in[1];  // [2,E] int64
    const float*     Wm   = (const float*)d_in[2];      // [F,F]
    const float*     attS = (const float*)d_in[3];      // [1,F]
    const float*     attD = (const float*)d_in[4];      // [1,F]
    const float*     bias = (const float*)d_in[5];      // [F]
    float*           out  = (float*)d_out;              // [B,N,F]

    // workspace layout (~53 MB)
    float*    H      = (float*)d_ws;                    // BN * F
    float*    asrc   = H + (size_t)BN_ * F_;            // BN
    float*    adst   = asrc + BN_;                      // BN
    unsigned* maxkey = (unsigned*)(adst + BN_);         // BN
    float*    denom  = (float*)(maxkey + BN_);          // BN

    const int TB = 256;
    // 1) init accumulators / bias
    gat_init<<<(BNF_ + TB - 1) / TB, TB, 0, stream>>>(out, bias, maxkey, denom);
    // 2) WMMA GEMM: 6250 row tiles = 1250 blocks * ROWT_; 8 waves = 8 col tiles
    gat_gemm_wmma<<<BN_ / (16 * ROWT_), TB, 0, stream>>>(x, Wm, H);
    // 3) attention coefficients (wave per row)
    gat_dots<<<((long long)BN_ * 32 + TB - 1) / TB, TB, 0, stream>>>(H, attS, attD, asrc, adst);
    // 4) segment max
    long long BEp = (long long)B_ * EP_;
    gat_emax<<<(BEp + TB - 1) / TB, TB, 0, stream>>>(ei, asrc, adst, maxkey);
    // 5) softmax denominator
    gat_esum<<<(BEp + TB - 1) / TB, TB, 0, stream>>>(ei, asrc, adst, maxkey, denom);
    // 6) weighted scatter (wave per (batch,edge))
    gat_scatter<<<(BEp * 32 + TB - 1) / TB, TB, 0, stream>>>(ei, H, asrc, adst, maxkey, denom, out);
}
